// Attention_17506286699073
// MI455X (gfx1250) — matrix operand order repaired and checked by reading, not compiled, not run on hardware
//
#include <hip/hip_runtime.h>
#include <hip/hip_bf16.h>

// Problem constants from the reference: B=32, H=64, W=64, C=256.
#define BATCH 32
#define HW    4096   // H*W, the contraction length for scores
#define CH    256    // channel dim (q and k axes of the score matrix)

typedef __attribute__((ext_vector_type(16))) __bf16 v16bf;
typedef __attribute__((ext_vector_type(8)))  float  v8f;

// A-fragment K index for v_wmma_*_16x16x32_bf16 (wave32):
// lanes 0-15 hold K = 0..7 (V0-3) and 16..23 (V4-7); lanes 16-31 hold 8..15 and 24..31.
__device__ __forceinline__ int ka_idx(int half, int e) {
    return half * 8 + ((e < 8) ? e : (8 + e));
}
// Contiguous K index (unused; B fragments use ka_idx): lanes 0-15 hold K = 0..15 (2 per VGPR), lanes 16-31 hold K = 16..31.
__device__ __forceinline__ int kb_idx(int half, int e) {
    return half * 16 + e;
}

// Pack 16 floats (4x float4) into a bf16 fragment, in order.
__device__ __forceinline__ v16bf pack16(float4 r0, float4 r1, float4 r2, float4 r3) {
    v16bf f;
    f[0]  = (__bf16)r0.x; f[1]  = (__bf16)r0.y; f[2]  = (__bf16)r0.z; f[3]  = (__bf16)r0.w;
    f[4]  = (__bf16)r1.x; f[5]  = (__bf16)r1.y; f[6]  = (__bf16)r1.z; f[7]  = (__bf16)r1.w;
    f[8]  = (__bf16)r2.x; f[9]  = (__bf16)r2.y; f[10] = (__bf16)r2.z; f[11] = (__bf16)r2.w;
    f[12] = (__bf16)r3.x; f[13] = (__bf16)r3.y; f[14] = (__bf16)r3.z; f[15] = (__bf16)r3.w;
    return f;
}

// -----------------------------------------------------------------------------
// Kernel 1: S[b, m, n] = sum_l Q[b, l, m] * K[b, l, n]        (m,n in [0,256))
// One wave computes a 32x32 tile of S (2x2 grid of 16x16 WMMA accumulators).
// Per-lane gathers are row-strided (lane = column index) but coalesced across
// lanes: each load instruction touches two 64B segments.
// -----------------------------------------------------------------------------
__global__ void attn_scores_kernel(const float* __restrict__ q,
                                   const float* __restrict__ k,
                                   float* __restrict__ S) {
    const int wave = (blockIdx.x * blockDim.x + threadIdx.x) >> 5;
    const int lane = threadIdx.x & 31;
    const int lm   = lane & 15;     // M (for A) / N (for B) index within tile
    const int half = lane >> 4;     // which K half this lane holds

    const int b  = wave >> 6;              // 64 tiles per batch
    const int t  = wave & 63;
    const int m0 = (t >> 3) << 5;          // 32-row block of q-axis
    const int n0 = (t & 7) << 5;           // 32-col block of k-axis

    const float* __restrict__ Qb = q + (size_t)b * HW * CH;
    const float* __restrict__ Kb = k + (size_t)b * HW * CH;

    v8f acc[2][2] = {};

    for (int l0 = 0; l0 < HW; l0 += 32) {
        // Prefetch next chunk's rows into cache (global_prefetch_b8).
        if (l0 + 32 < HW) {
            __builtin_prefetch(Qb + (size_t)(l0 + 32 + half * 16) * CH + m0 + lm, 0, 0);
            __builtin_prefetch(Kb + (size_t)(l0 + 32 + half * 16) * CH + n0 + lm, 0, 0);
        }

        v16bf af[2], bf[2];
#pragma unroll
        for (int e = 0; e < 16; ++e) {
            const int kk = ka_idx(half, e);
            const float* row = Qb + (size_t)(l0 + kk) * CH;   // Q[l, :] row
            af[0][e] = (__bf16)row[m0 + lm];                  // coalesced over lanes
            af[1][e] = (__bf16)row[m0 + 16 + lm];
        }
#pragma unroll
        for (int e = 0; e < 16; ++e) {
            const int kk = ka_idx(half, e);
            const float* row = Kb + (size_t)(l0 + kk) * CH;   // K[l, :] row
            bf[0][e] = (__bf16)row[n0 + lm];
            bf[1][e] = (__bf16)row[n0 + 16 + lm];
        }
#pragma unroll
        for (int i = 0; i < 2; ++i)
#pragma unroll
            for (int j = 0; j < 2; ++j)
                acc[i][j] = __builtin_amdgcn_wmma_f32_16x16x32_bf16(
                    false, af[i], false, bf[j], (short)0, acc[i][j], false, false);
    }

    // D layout: VGPR r -> M = r + 8*half, N = lm (per 16x16 tile)
#pragma unroll
    for (int i = 0; i < 2; ++i)
#pragma unroll
        for (int j = 0; j < 2; ++j)
#pragma unroll
            for (int r = 0; r < 8; ++r) {
                const int M = m0 + i * 16 + half * 8 + r;
                const int N = n0 + j * 16 + lm;
                S[((size_t)b * CH + M) * CH + N] = acc[i][j][r];
            }
}

// -----------------------------------------------------------------------------
// Kernel 2: numerically-stable softmax over the last axis, in place.
// One 256-thread block per (b, q) row of the 256-wide score row.
// -----------------------------------------------------------------------------
__global__ void attn_softmax_kernel(float* __restrict__ S) {
    const int row = blockIdx.x;       // b*256 + q
    const int tid = threadIdx.x;      // 0..255 == key index
    __shared__ float red[256];

    float v = S[(size_t)row * CH + tid];

    red[tid] = v;
    __syncthreads();
#pragma unroll
    for (int s = 128; s > 0; s >>= 1) {
        if (tid < s) red[tid] = fmaxf(red[tid], red[tid + s]);
        __syncthreads();
    }
    const float mx = red[0];
    __syncthreads();

    const float e = __expf(v - mx);
    red[tid] = e;
    __syncthreads();
#pragma unroll
    for (int s = 128; s > 0; s >>= 1) {
        if (tid < s) red[tid] += red[tid + s];
        __syncthreads();
    }
    S[(size_t)row * CH + tid] = e / red[0];
}

// -----------------------------------------------------------------------------
// Kernel 3: out[b, m, n] = sum_k P[b, m, k] * V[b, n, k]   (m=q in [0,256),
// n = h*W+w in [0,4096)).  N maps to the contiguous hw axis -> coalesced stores.
// One wave per 32x32 output tile.  Both operand streams are contiguous per
// lane along k, so all fragment loads are 16B global_load_b128.
// -----------------------------------------------------------------------------
__global__ void attn_out_kernel(const float* __restrict__ P,
                                const float* __restrict__ v,
                                float* __restrict__ out) {
    const int wave = (blockIdx.x * blockDim.x + threadIdx.x) >> 5;
    const int lane = threadIdx.x & 31;
    const int lm   = lane & 15;
    const int half = lane >> 4;

    const int b  = wave >> 10;             // 1024 tiles per batch
    const int t  = wave & 1023;
    const int m0 = (t >> 7) << 5;          // q-axis block (8 blocks)
    const int n0 = (t & 127) << 5;         // hw-axis block (128 blocks)

    const float* __restrict__ Pb = P + (size_t)b * CH * CH;
    const float* __restrict__ Vb = v + (size_t)b * HW * CH;

    v8f acc[2][2] = {};

    for (int k0 = 0; k0 < CH; k0 += 32) {
        v16bf af[2], bf[2];
        // A fragment: rows m, K pattern = two contiguous runs of 8 floats.
#pragma unroll
        for (int i = 0; i < 2; ++i) {
            const float* base = Pb + (size_t)(m0 + i * 16 + lm) * CH + k0 + half * 8;
            const float4 r0 = *(const float4*)(base);        // K = kb+0..3
            const float4 r1 = *(const float4*)(base + 4);    // K = kb+4..7
            const float4 r2 = *(const float4*)(base + 16);   // K = kb+16..19
            const float4 r3 = *(const float4*)(base + 20);   // K = kb+20..23
            af[i] = pack16(r0, r1, r2, r3);
        }
        // B fragment: B[k, n] = V[n, k]; same K pattern as A (two runs of 8 floats).
#pragma unroll
        for (int j = 0; j < 2; ++j) {
            const float* base = Vb + (size_t)(n0 + j * 16 + lm) * CH + k0 + half * 8;
            const float4 r0 = *(const float4*)(base);
            const float4 r1 = *(const float4*)(base + 4);
            const float4 r2 = *(const float4*)(base + 16);
            const float4 r3 = *(const float4*)(base + 20);
            bf[j] = pack16(r0, r1, r2, r3);
        }
#pragma unroll
        for (int i = 0; i < 2; ++i)
#pragma unroll
            for (int j = 0; j < 2; ++j)
                acc[i][j] = __builtin_amdgcn_wmma_f32_16x16x32_bf16(
                    false, af[i], false, bf[j], (short)0, acc[i][j], false, false);
    }

#pragma unroll
    for (int i = 0; i < 2; ++i)
#pragma unroll
        for (int j = 0; j < 2; ++j)
#pragma unroll
            for (int r = 0; r < 8; ++r) {
                const int M = m0 + i * 16 + half * 8 + r;
                const int N = n0 + j * 16 + lm;
                out[((size_t)b * CH + M) * HW + N] = acc[i][j][r];
            }
}

extern "C" void kernel_launch(void* const* d_in, const int* in_sizes, int n_in,
                              void* d_out, int out_size, void* d_ws, size_t ws_size,
                              hipStream_t stream) {
    (void)in_sizes; (void)n_in; (void)out_size; (void)ws_size;

    const float* q = (const float*)d_in[0];
    const float* k = (const float*)d_in[1];
    const float* v = (const float*)d_in[2];
    float* out = (float*)d_out;
    float* S   = (float*)d_ws;   // B*256*256 f32 = 8 MB of scratch

    // Kernel 1: 32 batches * 64 tiles = 2048 waves; 8 waves (256 thr) per block.
    attn_scores_kernel<<<2048 / 8, 256, 0, stream>>>(q, k, S);

    // Kernel 2: one block per score row (B*256 rows).
    attn_softmax_kernel<<<BATCH * CH, 256, 0, stream>>>(S);

    // Kernel 3: 32 batches * 1024 tiles = 32768 waves; 8 waves per block.
    attn_out_kernel<<<32768 / 8, 256, 0, stream>>>(S, v, out);
}
